// Quantize_61443802136632
// MI455X (gfx1250) — compile-verified, matrix operand order checked by reading, not run on hardware
//
#include <hip/hip_runtime.h>
#include <hip/hip_bf16.h>

#define SIZE 1024
#define CODE_DIM 256
#define NPTS 65536
#define NT (SIZE / 16)               // 64 code tiles
#define TILE_HALVES (16 * CODE_DIM)  // 4096 halves = 8 KB per tile

typedef __attribute__((ext_vector_type(16))) _Float16 v16h;
typedef __attribute__((ext_vector_type(8)))  _Float16 v8h;
typedef __attribute__((ext_vector_type(8)))  float    v8f;
typedef __attribute__((ext_vector_type(4)))  float    v4f;
typedef __attribute__((ext_vector_type(4)))  int      v4i;

// ---------------------------------------------------------------------------
// Kernel 1: codebook prep.  weight f32 -> f16 and 0.5*|w|^2 per code.
// ---------------------------------------------------------------------------
__global__ void vq_prep_kernel(const float* __restrict__ weight,
                               _Float16* __restrict__ wf16,
                               float* __restrict__ whalf) {
  int wave = (blockIdx.x * blockDim.x + threadIdx.x) >> 5;
  int lane = threadIdx.x & 31;
  if (wave >= SIZE) return;
  const float* wr = weight + wave * CODE_DIM;
  const v4f* p = (const v4f*)(wr + lane * 8);
  v4f f0 = p[0], f1 = p[1];
  v8h h;
  float ss = 0.f;
#pragma unroll
  for (int i = 0; i < 4; ++i) { h[i]     = (_Float16)f0[i]; ss += f0[i] * f0[i]; }
#pragma unroll
  for (int i = 0; i < 4; ++i) { h[4 + i] = (_Float16)f1[i]; ss += f1[i] * f1[i]; }
  *(v8h*)(wf16 + wave * CODE_DIM + lane * 8) = h;
#pragma unroll
  for (int m = 16; m >= 1; m >>= 1) ss += __shfl_xor(ss, m, 32);
  if (lane == 0) whalf[wave] = 0.5f * ss;
}

// ---------------------------------------------------------------------------
// Async staging helpers (CDNA5 global_load_async_to_lds + ASYNCcnt), with
// a synchronous ds_store fallback if the builtins aren't available.
// ---------------------------------------------------------------------------
#if __has_builtin(__builtin_amdgcn_global_load_async_to_lds_b128)
#define VQ_ASYNC_STAGE 1
#else
#define VQ_ASYNC_STAGE 0
#endif

template <int N>
__device__ __forceinline__ void vq_wait_asynccnt() {
#if __has_builtin(__builtin_amdgcn_s_wait_asynccnt)
  __builtin_amdgcn_s_wait_asynccnt(N);
#else
  asm volatile("s_wait_asynccnt %0" ::"i"(N) : "memory");
#endif
}

// Copy one 8 KB code tile (16 codes x 256 halves) into LDS, spread over the
// 256 threads of the block: 2 x 16-byte chunks per thread.
__device__ __forceinline__ void vq_stage_tile(const _Float16* __restrict__ wf16,
                                              _Float16* __restrict__ dst,
                                              int tile, int t) {
  const _Float16* src = wf16 + (size_t)tile * TILE_HALVES;
#pragma unroll
  for (int i = 0; i < 2; ++i) {
    int c = t + i * 256;              // chunk id, 512 x 16B chunks per tile
#if VQ_ASYNC_STAGE
    v4i* gsrc_gen = (v4i*)const_cast<_Float16*>(src + c * 8);
    v4i* ldst_gen = (v4i*)(dst + c * 8);
    __builtin_amdgcn_global_load_async_to_lds_b128(
        (__attribute__((address_space(1))) v4i*)gsrc_gen,
        (__attribute__((address_space(3))) v4i*)ldst_gen,
        0, 0);
#else
    v8h v = *(const v8h*)(src + c * 8);
    *(v8h*)(dst + c * 8) = v;
#endif
  }
}

// Load one 16x32 B chunk (ISA 16-bit B-matrix layout) from the LDS tile.
__device__ __forceinline__ v16h vq_ldB(const _Float16* __restrict__ bs,
                                       int kc, int hi8) {
  const int kb = kc * 32;
  v8h b0 = *(const v8h*)(bs + kb + hi8);        // ds_load_b128
  v8h b1 = *(const v8h*)(bs + kb + 16 + hi8);   // ds_load_b128
  return __builtin_shufflevector(b0, b1, 0, 1, 2, 3, 4, 5, 6, 7,
                                         8, 9, 10, 11, 12, 13, 14, 15);
}

// ---------------------------------------------------------------------------
// Kernel 2: WMMA argmin.  8 waves x 16 rows per block.  A held in registers;
// B double-buffered through LDS with async copies shared by all 8 waves and
// software-pipelined (chunk kc+1 loads overlap the WMMA of chunk kc).
// Score = z.w - 0.5|w|^2; running argmax per row; shfl_xor final reduction.
// ---------------------------------------------------------------------------
__global__ __launch_bounds__(256, 1)
void vq_argmin_kernel(const float* __restrict__ z,
                      const _Float16* __restrict__ wf16,
                      const float* __restrict__ whalf,
                      int* __restrict__ idx_out,
                      float* __restrict__ idx_f_out) {
  __shared__ _Float16 sB[2][TILE_HALVES];   // 2 x 8 KB double buffer
  __shared__ float    sWn[SIZE];            // 4 KB: 0.5*|w|^2 per code

  const int t       = threadIdx.x;
  const int lane    = t & 31;
  const int wave    = t >> 5;
  const int rowBase = (blockIdx.x * 8 + wave) * 16;
  const int col16   = lane & 15;          // column / row-within-tile id
  const int hi8     = (lane >> 4) * 8;    // K sub-offset (A/B), row offset (C)

  // Stage all code norms into LDS once (covered by the first loop barrier).
  {
    v4f w4 = *(const v4f*)(whalf + t * 4);
    *(v4f*)(sWn + t * 4) = w4;
  }

  // ---- Load A: 16 rows x 256 K, f32 -> f16, ISA 16-bit A-matrix layout ----
  v16h a[8];
  const float* zr = z + (size_t)(rowBase + col16) * CODE_DIM;
#pragma unroll
  for (int kc = 0; kc < 8; ++kc) {
    const int kb = kc * 32;
    const v4f* p0 = (const v4f*)(zr + kb + hi8);
    const v4f* p1 = (const v4f*)(zr + kb + 16 + hi8);
    v4f f0 = p0[0], f1 = p0[1], f2 = p1[0], f3 = p1[1];
    v16h tt;
#pragma unroll
    for (int i = 0; i < 4; ++i) {
      tt[i]      = (_Float16)f0[i];
      tt[4 + i]  = (_Float16)f1[i];
      tt[8 + i]  = (_Float16)f2[i];
      tt[12 + i] = (_Float16)f3[i];
    }
    a[kc] = tt;
  }

  float bestVal[8];
  int   bestIdx[8];
#pragma unroll
  for (int r = 0; r < 8; ++r) { bestVal[r] = -3.0e38f; bestIdx[r] = 0; }

  // Prologue: stage tile 0 into buffer 0.
  vq_stage_tile(wf16, &sB[0][0], 0, t);

  for (int j = 0; j < NT; ++j) {
    const int cur = j & 1;
    if (j + 1 < NT) {
      // Issue next tile into the other buffer (its previous consumers
      // finished at the end-of-iteration barrier of iteration j-1).
      vq_stage_tile(wf16, &sB[cur ^ 1][0], j + 1, t);
      vq_wait_asynccnt<2>();   // tile j's 2 per-thread async ops retired
    } else {
      vq_wait_asynccnt<0>();
    }
    __syncthreads();           // tile j visible to all waves

    const int c = j * 16 + col16;                 // this lane's code column
    const _Float16* bs = &sB[cur][col16 * CODE_DIM];

    // Software-pipelined K loop: load chunk kc+1 while WMMA kc executes.
    v16h bcur = vq_ldB(bs, 0, hi8);
    v8f acc = {};
#pragma unroll
    for (int kc = 0; kc < 8; ++kc) {
      v16h bnext = bcur;
      if (kc < 7) bnext = vq_ldB(bs, kc + 1, hi8);
      acc = __builtin_amdgcn_wmma_f32_16x16x32_f16(
          /*neg_a=*/false, a[kc], /*neg_b=*/false, bcur,
          /*c_mod=*/(short)0, acc, /*reuse_a=*/false, /*reuse_b=*/false);
      bcur = bnext;
    }

    const float wn = sWn[c];                      // ds_load_b32, short latency
#pragma unroll
    for (int r = 0; r < 8; ++r) {
      float s = acc[r] - wn;
      if (s > bestVal[r]) { bestVal[r] = s; bestIdx[r] = c; }
    }
    __syncthreads();           // done with sB[cur]; safe to overwrite next iter
  }

  // Argmax across the 16 lanes of each half-wave (masks 1..8 stay in-group).
  // C layout: VGPR r = row (rowBase + r) on lanes 0-15, row (rowBase+8+r)
  // on lanes 16-31.  Ties prefer the lower code index (matches jnp.argmax).
#pragma unroll
  for (int m = 1; m <= 8; m <<= 1) {
#pragma unroll
    for (int r = 0; r < 8; ++r) {
      float ov = __shfl_xor(bestVal[r], m, 32);
      int   oi = __shfl_xor(bestIdx[r], m, 32);
      if (ov > bestVal[r] || (ov == bestVal[r] && oi < bestIdx[r])) {
        bestVal[r] = ov; bestIdx[r] = oi;
      }
    }
  }

  if (col16 == 0) {
    const int rb = rowBase + hi8;   // lane 0 -> rows 0..7, lane 16 -> rows 8..15
#pragma unroll
    for (int r = 0; r < 8; ++r) {
      idx_out[rb + r]   = bestIdx[r];
      idx_f_out[rb + r] = (float)bestIdx[r];
    }
  }
}

// ---------------------------------------------------------------------------
// Kernel 3: gather.  quantized[n] = weight[idx[n]].  One wave per row,
// 2 x B128 load/store per lane; codebook is L2-hot -> store-bandwidth bound.
// ---------------------------------------------------------------------------
__global__ void vq_gather_kernel(const float* __restrict__ weight,
                                 const int* __restrict__ idx,
                                 float* __restrict__ out) {
  int wave = (blockIdx.x * blockDim.x + threadIdx.x) >> 5;
  int lane = threadIdx.x & 31;
  if (wave >= NPTS) return;
  int c = idx[wave];
  const v4f* src = (const v4f*)(weight + (size_t)c * CODE_DIM + lane * 8);
  v4f lo = src[0], hi = src[1];
  v4f* dst = (v4f*)(out + (size_t)wave * CODE_DIM + lane * 8);
  dst[0] = lo;
  dst[1] = hi;
}

// ---------------------------------------------------------------------------
extern "C" void kernel_launch(void* const* d_in, const int* in_sizes, int n_in,
                              void* d_out, int out_size, void* d_ws, size_t ws_size,
                              hipStream_t stream) {
  const float* z      = (const float*)d_in[0];   // [N, 256] fp32
  const float* weight = (const float*)d_in[1];   // [1024, 256] fp32

  float* out_q   = (float*)d_out;                      // [N, 256]
  float* out_idx = out_q + (size_t)NPTS * CODE_DIM;    // [N] (indices as float)

  // Workspace: f16 codebook (512 KB) | 0.5*|w|^2 (4 KB) | int idx (256 KB)
  char* ws = (char*)d_ws;
  _Float16* wf16 = (_Float16*)ws;
  float* whalf   = (float*)(ws + (size_t)SIZE * CODE_DIM * 2);
  int* idxs      = (int*)(ws + (size_t)SIZE * CODE_DIM * 2 + SIZE * 4);

  // 1) codebook prep: 1024 waves (8 per block)
  vq_prep_kernel<<<SIZE / 8, 256, 0, stream>>>(weight, wf16, whalf);
  // 2) WMMA argmin: 512 blocks x 8 waves, 16 rows per wave
  vq_argmin_kernel<<<NPTS / 128, 256, 0, stream>>>(z, wf16, whalf, idxs, out_idx);
  // 3) gather: 65536 waves (8 per block)
  vq_gather_kernel<<<NPTS / 8, 256, 0, stream>>>(weight, idxs, out_q);
}
